// RecurrentEncoder_56873956933919
// MI455X (gfx1250) — matrix operand order repaired and checked by reading, not compiled, not run on hardware
//
#include <hip/hip_runtime.h>
#include <hip/hip_bf16.h>
#include <stdint.h>

#define B_TOT   4096
#define T_TOT   32
#define IN_DIM  784
#define IN_PAD  800          // 784 padded to multiple of 32 (WMMA K)
#define HD      1024
#define KP      5            // NPARAMS
#define BM      32           // batch rows per block
#define NBLK    (B_TOT / BM) // 128 blocks
#define BN      256          // N strip per pass
#define BK      128          // K depth per LDS stage
#define BSTE    (BN * BK)    // elements per stage buffer

typedef _Float16 v16h __attribute__((ext_vector_type(16)));
typedef _Float16 v8h  __attribute__((ext_vector_type(8)));
typedef float    v8f  __attribute__((ext_vector_type(8)));

union V16 { v16h v; v8h h[2]; };

__device__ __forceinline__ float softplusf(float x) {
    return fmaxf(x, 0.f) + log1pf(expf(-fabsf(x)));
}

// Issue one 256(N) x 128(K) fp16 W_hh stage into LDS via async-to-LDS DMA.
// Each thread owns one N row; 16 x b128 per thread (16 async ops per wave).
// The instruction offset applies to BOTH lds and global addresses (ISA 10.x).
__device__ __forceinline__ void issue_b_stage(const _Float16* __restrict__ Whh_h,
                                              _Float16* bst_buf,
                                              int nbase, int kb, int tid) {
    const uint64_t g = (uint64_t)(uintptr_t)(Whh_h + (size_t)(nbase + tid) * HD + kb);
    const uint32_t l = (uint32_t)(uintptr_t)(bst_buf + (size_t)tid * BK);
#pragma unroll
    for (int kc = 0; kc < 16; ++kc) {
        asm volatile("global_load_async_to_lds_b128 %0, %1, off offset:%2"
                     :: "v"(l), "v"(g), "i"(kc * 16) : "memory");
    }
}

// ---------------------------------------------------------------------------
// Kernel A: fp32 -> fp16 conversion (with K padding for x / W_ih)
// ---------------------------------------------------------------------------
__global__ __launch_bounds__(256) void convert_kernel(
        const float* __restrict__ x,
        const float* __restrict__ W_ih,
        const float* __restrict__ W_hh,
        _Float16* __restrict__ x_h,
        _Float16* __restrict__ Wih_h,
        _Float16* __restrict__ Whh_h) {
    const int stride = gridDim.x * blockDim.x;
    const int tid = blockIdx.x * blockDim.x + threadIdx.x;
    for (int i = tid; i < B_TOT * IN_PAD; i += stride) {
        const int r = i / IN_PAD, c = i % IN_PAD;
        x_h[i] = (c < IN_DIM) ? (_Float16)x[r * IN_DIM + c] : (_Float16)0.f;
    }
    for (int i = tid; i < HD * IN_PAD; i += stride) {
        const int r = i / IN_PAD, c = i % IN_PAD;
        Wih_h[i] = (c < IN_DIM) ? (_Float16)W_ih[r * IN_DIM + c] : (_Float16)0.f;
    }
    for (int i = tid; i < HD * HD; i += stride) {
        Whh_h[i] = (_Float16)W_hh[i];
    }
}

// ---------------------------------------------------------------------------
// Kernel B: x_proj = x @ W_ih^T + b_ih (one-shot, direct-from-L2 operands)
// ---------------------------------------------------------------------------
__global__ __launch_bounds__(256) void xproj_kernel(
        const _Float16* __restrict__ x_h,
        const _Float16* __restrict__ Wih_h,
        const float* __restrict__ b_ih,
        float* __restrict__ xp) {
    const int lane = threadIdx.x & 31;
    const int wave = threadIdx.x >> 5;
    const int mw   = wave & 1;
    const int nw   = wave >> 1;
    const int rb   = blockIdx.x * BM;
    const int lr   = lane & 15;
    const int lh   = lane >> 4;

    const int arow = rb + mw * 16 + lr;
    for (int p = 0; p < 4; ++p) {
        const int n0 = p * 256 + nw * 64;
        v8f acc[4] = {};
        for (int k0 = 0; k0 < IN_PAD; k0 += 32) {
            V16 a;
            const _Float16* ap = x_h + arow * IN_PAD + k0 + lh * 8;
            a.h[0] = *(const v8h*)(ap);
            a.h[1] = *(const v8h*)(ap + 16);
#pragma unroll
            for (int tn = 0; tn < 4; ++tn) {
                const int bcol = n0 + tn * 16 + lr;
                const _Float16* bp = Wih_h + bcol * IN_PAD + k0 + lh * 8;
                V16 b;
                b.h[0] = *(const v8h*)(bp);
                b.h[1] = *(const v8h*)(bp + 16);
                acc[tn] = __builtin_amdgcn_wmma_f32_16x16x32_f16(
                    false, a.v, false, b.v, (short)0, acc[tn], false, false);
            }
        }
#pragma unroll
        for (int tn = 0; tn < 4; ++tn) {
            const int n = n0 + tn * 16 + lr;
            const float bias = b_ih[n];
            const int mbase = rb + mw * 16 + lh * 8;
#pragma unroll
            for (int r = 0; r < 8; ++r) {
                xp[(mbase + r) * HD + n] = acc[tn][r] + bias;
            }
        }
    }
}

// ---------------------------------------------------------------------------
// Kernel C: persistent recurrence. 32 batch rows per block; h ping-pongs in
// LDS; W_hh strip double-buffered in LDS via global_load_async_to_lds_b128.
// ---------------------------------------------------------------------------
__global__ __launch_bounds__(256) void recur_kernel(
        const float* __restrict__ hx,
        const float* __restrict__ eps,
        const float* __restrict__ b_hh,
        const float* __restrict__ fe_W,
        const float* __restrict__ fe_b,
        const _Float16* __restrict__ Whh_h,
        const float* __restrict__ xp,
        float* __restrict__ out) {
    extern __shared__ char smem_raw[];
    _Float16* hbuf = (_Float16*)smem_raw;                              // 2*BM*HD
    _Float16* bst  = (_Float16*)(smem_raw + 2 * BM * HD * 2);          // 2*BSTE
    float* fe_s    = (float*)(smem_raw + 2 * BM * HD * 2 + 2 * BSTE * 2);

    const int tid  = threadIdx.x;
    const int lane = tid & 31;
    const int wave = tid >> 5;
    const int mw   = wave & 1;
    const int nw   = wave >> 1;
    const int rb   = blockIdx.x * BM;
    const int lr   = lane & 15;
    const int lh   = lane >> 4;

    for (int i = tid; i < BM * HD; i += 256) {
        hbuf[i] = (_Float16)hx[(rb + i / HD) * HD + (i % HD)];
    }
    __syncthreads();

    int cur = 0;
    for (int t = 0; t < T_TOT; ++t) {
        const _Float16* hin = hbuf + cur * BM * HD;
        _Float16* hout = hbuf + (1 - cur) * BM * HD;

        for (int p = 0; p < 4; ++p) {
            const int nbase = p * BN;
            v8f acc[4] = {};

            issue_b_stage(Whh_h, bst, nbase, 0, tid);   // prologue stage 0
            int sb = 0;
            for (int kb = 0; kb < HD; kb += BK) {
                if (kb + BK < HD) {
                    issue_b_stage(Whh_h, bst + (sb ^ 1) * BSTE, nbase, kb + BK, tid);
                    asm volatile("s_wait_asynccnt 16" ::: "memory");
                } else {
                    asm volatile("s_wait_asynccnt 0" ::: "memory");
                }
                __syncthreads();

                const _Float16* bs = bst + sb * BSTE;
#pragma unroll
                for (int ks = 0; ks < BK; ks += 32) {
                    V16 a;
                    const _Float16* ap = hin + (mw * 16 + lr) * HD + kb + ks + lh * 8;
                    a.h[0] = *(const v8h*)(ap);
                    a.h[1] = *(const v8h*)(ap + 16);
#pragma unroll
                    for (int tn = 0; tn < 4; ++tn) {
                        const int c = nw * 64 + tn * 16 + lr;
                        const _Float16* bp = bs + c * BK + ks + lh * 8;
                        V16 b;
                        b.h[0] = *(const v8h*)(bp);
                        b.h[1] = *(const v8h*)(bp + 16);
                        acc[tn] = __builtin_amdgcn_wmma_f32_16x16x32_f16(
                            false, a.v, false, b.v, (short)0, acc[tn], false, false);
                    }
                }
                __syncthreads();   // protects stage buffer reuse
                sb ^= 1;
            }

            // epilogue: h_new = tanh(acc + xp + b_hh)
#pragma unroll
            for (int tn = 0; tn < 4; ++tn) {
                const int n = nbase + nw * 64 + tn * 16 + lr;
                const float bias = b_hh[n];
                const int mloc = mw * 16 + lh * 8;
#pragma unroll
                for (int r = 0; r < 8; ++r) {
                    const float v = acc[tn][r] + xp[(rb + mloc + r) * HD + n] + bias;
                    hout[(mloc + r) * HD + n] = (_Float16)tanhf(v);
                }
            }
        }
        __syncthreads();

        // fe = h @ fe_W^T + fe_b : BM x 5 head
        if (tid < BM * KP) {
            const int row = tid / KP;
            const int pidx = tid % KP;
            const _Float16* hr = hout + row * HD;
            const float* wr = fe_W + pidx * HD;
            float dot = 0.f;
            for (int k = 0; k < HD; k += 4) {
                dot += (float)hr[k]     * wr[k]
                     + (float)hr[k + 1] * wr[k + 1]
                     + (float)hr[k + 2] * wr[k + 2]
                     + (float)hr[k + 3] * wr[k + 3];
            }
            fe_s[tid] = dot + fe_b[pidx];
        }
        __syncthreads();

        if (tid < BM) {
            const int b = rb + tid;
            const float mu0 = fe_s[tid * KP + 0];
            const float mu1 = fe_s[tid * KP + 1];
            const float L00 = softplusf(fe_s[tid * KP + 2]);
            const float L10 = fe_s[tid * KP + 3];
            const float L11 = softplusf(fe_s[tid * KP + 4]);
            const float e0 = eps[(t * B_TOT + b) * 2 + 0];
            const float e1 = eps[(t * B_TOT + b) * 2 + 1];
            const float z0 = mu0 + L00 * e0;
            const float z1 = mu1 + L10 * e0 + L11 * e1;
            const int ZS = B_TOT * 2 * T_TOT;        // 262144
            out[b * 2 * T_TOT + t]              = z0;
            out[b * 2 * T_TOT + T_TOT + t]      = z1;
            out[ZS + b * 2 * T_TOT + t]         = z0;
            out[ZS + b * 2 * T_TOT + T_TOT + t] = z1;
            float* Lo = out + 2 * ZS + b * 4 * T_TOT;
            Lo[0 * T_TOT + t] = L00;
            Lo[1 * T_TOT + t] = 0.f;
            Lo[2 * T_TOT + t] = L10;
            Lo[3 * T_TOT + t] = L11;
        }
        __syncthreads();
        cur ^= 1;
    }
}

// ---------------------------------------------------------------------------
extern "C" void kernel_launch(void* const* d_in, const int* in_sizes, int n_in,
                              void* d_out, int out_size, void* d_ws, size_t ws_size,
                              hipStream_t stream) {
    (void)in_sizes; (void)n_in; (void)out_size; (void)ws_size;
    const float* x    = (const float*)d_in[0];
    const float* hx   = (const float*)d_in[1];
    const float* eps  = (const float*)d_in[2];
    const float* W_ih = (const float*)d_in[3];
    const float* b_ih = (const float*)d_in[4];
    const float* W_hh = (const float*)d_in[5];
    const float* b_hh = (const float*)d_in[6];
    const float* fe_W = (const float*)d_in[7];
    const float* fe_b = (const float*)d_in[8];
    float* out = (float*)d_out;

    char* ws = (char*)d_ws;
    float*    xp    = (float*)ws;                         // 4096*1024 f32 = 16 MiB
    _Float16* x_h   = (_Float16*)(ws + 16777216);         // 4096*800  f16
    _Float16* Wih_h = (_Float16*)(ws + 23330816);         // 1024*800  f16
    _Float16* Whh_h = (_Float16*)(ws + 24969216);         // 1024*1024 f16

    convert_kernel<<<512, 256, 0, stream>>>(x, W_ih, W_hh, x_h, Wih_h, Whh_h);
    xproj_kernel<<<NBLK, 256, 0, stream>>>(x_h, Wih_h, b_ih, xp);

    const size_t smem = 2 * BM * HD * sizeof(_Float16)       // h ping-pong
                      + 2 * BSTE * sizeof(_Float16)          // W_hh stage x2
                      + BM * KP * sizeof(float);             // fe scratch
    hipFuncSetAttribute((const void*)recur_kernel,
                        hipFuncAttributeMaxDynamicSharedMemorySize, (int)smem);
    recur_kernel<<<NBLK, 256, smem, stream>>>(hx, eps, b_hh, fe_W, fe_b,
                                              Whh_h, xp, out);
}
